// MemoryBankModel_7069516169476
// MI455X (gfx1250) — matrix operand order repaired and checked by reading, not compiled, not run on hardware
//
#include <hip/hip_runtime.h>
#include <math.h>

#define N_DIM 8192
#define M_DIM 50000
#define D_DIM 768
#define K_TOP 9
#define SEG   8
#define KSTEPS (D_DIM / 32)          // 24 k-steps of 32
#define NCHUNK (M_DIM / 16)          // 3125 chunks of 16 memory rows (exact)
#define NSLOTS ((size_t)NCHUNK * KSTEPS * 32)   // 2,400,000 operand slots

typedef __attribute__((ext_vector_type(16))) __bf16 v16bf;
typedef __attribute__((ext_vector_type(8)))  float  v8f;

union BF16x16 { unsigned u[8]; uint4 q[2]; v16bf v; };

// Split two fp32 (even/odd K) into packed bf16 hi and bf16 lo residual words.
__device__ __forceinline__ void split2(float a, float b, unsigned &h, unsigned &l) {
  unsigned ab = __float_as_uint(a), bb = __float_as_uint(b);
  unsigned ah = ab & 0xFFFF0000u,   bh = bb & 0xFFFF0000u;
  h = (ah >> 16) | bh;
  float ar = a - __uint_as_float(ah);
  float br = b - __uint_as_float(bh);
  l = (__float_as_uint(ar) >> 16) | (__float_as_uint(br) & 0xFFFF0000u);
}

__device__ __forceinline__ void topk_insert(float (&d)[K_TOP], int (&ix)[K_TOP],
                                            float sc, int id) {
  if (sc >= d[K_TOP - 1]) return;
  d[K_TOP - 1] = sc; ix[K_TOP - 1] = id;
  #pragma unroll
  for (int t = K_TOP - 1; t >= 1; --t) {
    if (d[t] < d[t - 1]) {
      float td = d[t]; d[t] = d[t - 1]; d[t - 1] = td;
      int   ti = ix[t]; ix[t] = ix[t - 1]; ix[t - 1] = ti;
    }
  }
}

// ---------------------------------------------------------------------------
// Kernel 0: row sum-of-squares for memory (m2) and query (q2). One wave/row.
// ---------------------------------------------------------------------------
__global__ __launch_bounds__(256) void rownorm_kernel(
    const float* __restrict__ query, const float* __restrict__ memory,
    float* __restrict__ m2, float* __restrict__ q2) {
  int row  = blockIdx.x * 8 + (threadIdx.x >> 5);
  int lane = threadIdx.x & 31;
  if (row >= M_DIM + N_DIM) return;
  const float* src = (row < M_DIM) ? (memory + (size_t)row * D_DIM)
                                   : (query + (size_t)(row - M_DIM) * D_DIM);
  float s = 0.0f;
  #pragma unroll 4
  for (int j = lane; j < D_DIM; j += 32) { float x = src[j]; s = fmaf(x, x, s); }
  #pragma unroll
  for (int off = 16; off > 0; off >>= 1) s += __shfl_xor(s, off, 32);
  if (lane == 0) {
    if (row < M_DIM) m2[row] = s;
    else             q2[row - M_DIM] = s;
  }
}

// ---------------------------------------------------------------------------
// Kernel 0b (fast path): one-shot operandization of the memory matrix into
// WMMA A-fragment order, bf16 hi + bf16 lo. Slot s = (chunk*KSTEPS+ks)*32+l:
// lane l's 16 bf16 for that k-step, 32 contiguous bytes -> coalesced loads
// in the GEMM loop.
// ---------------------------------------------------------------------------
__global__ __launch_bounds__(256) void operandize_kernel(
    const float* __restrict__ memory,
    unsigned* __restrict__ Ah, unsigned* __restrict__ Al) {
  size_t s = (size_t)blockIdx.x * 256 + threadIdx.x;
  if (s >= NSLOTS) return;
  int l     = (int)(s & 31);
  int ks    = (int)((s >> 5) % KSTEPS);
  int chunk = (int)(s / (KSTEPS * 32));
  int row   = chunk * 16 + (l & 15);
  int run0  = (l >> 4) * 8;               // lanes 0-15: K 0..7 & 16..23 ; 16-31: 8..15 & 24..31
  const float* src = memory + (size_t)row * D_DIM + ks * 32 + run0;
  float4 a0 = *(const float4*)(src + 0);
  float4 a1 = *(const float4*)(src + 4);
  float4 b0 = *(const float4*)(src + 16);
  float4 b1 = *(const float4*)(src + 20);
  unsigned* dh = Ah + s * 8;
  unsigned* dl = Al + s * 8;
  split2(a0.x, a0.y, dh[0], dl[0]);
  split2(a0.z, a0.w, dh[1], dl[1]);
  split2(a1.x, a1.y, dh[2], dl[2]);
  split2(a1.z, a1.w, dh[3], dl[3]);
  split2(b0.x, b0.y, dh[4], dl[4]);
  split2(b0.z, b0.w, dh[5], dl[5]);
  split2(b1.x, b1.y, dh[6], dl[6]);
  split2(b1.z, b1.w, dh[7], dl[7]);
}

// ---------------------------------------------------------------------------
// Shared B-staging: query tile (16 rows) -> LDS in exact WMMA-B operand order.
// ---------------------------------------------------------------------------
__device__ __forceinline__ void stage_query_tile(
    const float* __restrict__ query, int qbase, int tid,
    unsigned* ldsqh, unsigned* ldsql) {
  for (int slot = tid; slot < KSTEPS * 32; slot += 128) {
    int ks = slot >> 5, l = slot & 31;
    int q = l & 15, koff = (l >> 4) * 8;
    const float* src = query + (size_t)(qbase + q) * D_DIM + ks * 32 + koff;
    float4 f0 = *(const float4*)(src + 0);
    float4 f1 = *(const float4*)(src + 4);
    float4 f2 = *(const float4*)(src + 16);
    float4 f3 = *(const float4*)(src + 20);
    unsigned* dh = &ldsqh[slot * 8]; unsigned* dl = &ldsql[slot * 8];
    split2(f0.x, f0.y, dh[0], dl[0]);
    split2(f0.z, f0.w, dh[1], dl[1]);
    split2(f1.x, f1.y, dh[2], dl[2]);
    split2(f1.z, f1.w, dh[3], dl[3]);
    split2(f2.x, f2.y, dh[4], dl[4]);
    split2(f2.z, f2.w, dh[5], dl[5]);
    split2(f3.x, f3.y, dh[6], dl[6]);
    split2(f3.z, f3.w, dh[7], dl[7]);
  }
}

// ---------------------------------------------------------------------------
// Per-WG merge of 8 partial lists per query column + write to ws partials.
// ---------------------------------------------------------------------------
__device__ __forceinline__ void merge_and_write(
    int tid, int qbase, int seg, const float (&dl9)[K_TOP], const int (&il9)[K_TOP],
    float* md, int* mi, float* __restrict__ pd, int* __restrict__ pi) {
  #pragma unroll
  for (int j = 0; j < K_TOP; ++j) { md[tid * K_TOP + j] = dl9[j]; mi[tid * K_TOP + j] = il9[j]; }
  __syncthreads();
  if (tid < 16) {
    const float INF = __builtin_inff();
    float fd[K_TOP]; int fi[K_TOP];
    #pragma unroll
    for (int j = 0; j < K_TOP; ++j) { fd[j] = INF; fi[j] = -1; }
    for (int src = tid; src < 128; src += 16) {
      #pragma unroll
      for (int j = 0; j < K_TOP; ++j)
        topk_insert(fd, fi, md[src * K_TOP + j], mi[src * K_TOP + j]);
    }
    size_t o = ((size_t)(qbase + tid) * SEG + seg) * K_TOP;
    #pragma unroll
    for (int j = 0; j < K_TOP; ++j) { pd[o + j] = fd[j]; pi[o + j] = fi[j]; }
  }
}

// ---------------------------------------------------------------------------
// Kernel 1 (FAST): WMMA GEMM from pre-operandized bf16 hi/lo + per-lane top-9.
// Each wave: 2 chunks (32 memory rows) per iteration, one B fetch shared.
// Inner loop: 8 global_load_b128 + 4 ds_load_b128 + 6 wmma, ~no VALU.
// Grid: (N/16, SEG). Block 128.
// ---------------------------------------------------------------------------
__global__ __launch_bounds__(128) void knn_fast_kernel(
    const float* __restrict__ query,
    const unsigned* __restrict__ Ah, const unsigned* __restrict__ Al,
    const float* __restrict__ m2, float* __restrict__ pd, int* __restrict__ pi) {
  __shared__ unsigned ldsqh[KSTEPS * 32 * 8];   // 24 KB
  __shared__ unsigned ldsql[KSTEPS * 32 * 8];   // 24 KB
  __shared__ float    md[128 * K_TOP];
  __shared__ int      mi[128 * K_TOP];

  const int tid   = threadIdx.x;
  const int lane  = tid & 31;
  const int wave  = tid >> 5;
  const int qbase = blockIdx.x * 16;
  const int seg   = blockIdx.y;

  stage_query_tile(query, qbase, tid, ldsqh, ldsql);
  __syncthreads();

  const int CPS = (NCHUNK + SEG - 1) / SEG;     // 391 chunks per segment
  const int c0  = seg * CPS;
  const int c1  = (c0 + CPS < NCHUNK) ? (c0 + CPS) : NCHUNK;

  const float INF = __builtin_inff();
  float dl9[K_TOP]; int il9[K_TOP];
  #pragma unroll
  for (int j = 0; j < K_TOP; ++j) { dl9[j] = INF; il9[j] = -1; }

  const int laneU4 = lane * 2;                  // lane offset in uint4 units (8 uints)
  const int hi8    = (lane >> 4) << 3;          // C rows: lanes 0-15 -> +0, 16-31 -> +8

  for (int c = c0 + wave * 2; c < c1; c += 8) {
    const int  cB   = (c + 1 < c1) ? (c + 1) : c;
    const bool twoOk = (c + 1 < c1);

    const uint4* a1h = (const uint4*)Ah + ((size_t)c  * KSTEPS * 32) * 2 + laneU4;
    const uint4* a1l = (const uint4*)Al + ((size_t)c  * KSTEPS * 32) * 2 + laneU4;
    const uint4* a2h = (const uint4*)Ah + ((size_t)cB * KSTEPS * 32) * 2 + laneU4;
    const uint4* a2l = (const uint4*)Al + ((size_t)cB * KSTEPS * 32) * 2 + laneU4;

    if (c + 8 < c1)   // prefetch next pair's first lines
      __builtin_prefetch((const uint4*)Ah + ((size_t)(c + 8) * KSTEPS * 32) * 2 + laneU4, 0, 1);

    v8f acc1 = {}, acc2 = {};
    #pragma unroll 2
    for (int ks = 0; ks < KSTEPS; ++ks) {
      const int o = ks * 64;                    // 32 slots * 8 uints = 64 uint4 per k-step
      BF16x16 A1h, A1l, A2h, A2l, Bh, Bl;
      A1h.q[0] = a1h[o];  A1h.q[1] = a1h[o + 1];
      A1l.q[0] = a1l[o];  A1l.q[1] = a1l[o + 1];
      A2h.q[0] = a2h[o];  A2h.q[1] = a2h[o + 1];
      A2l.q[0] = a2l[o];  A2l.q[1] = a2l[o + 1];
      const unsigned* qh = &ldsqh[(ks * 32 + lane) * 8];
      const unsigned* ql = &ldsql[(ks * 32 + lane) * 8];
      Bh.q[0] = *(const uint4*)(qh + 0);  Bh.q[1] = *(const uint4*)(qh + 4);
      Bl.q[0] = *(const uint4*)(ql + 0);  Bl.q[1] = *(const uint4*)(ql + 4);
      // bf16x3 split: (mh+ml)(qh+ql) ~= mh*qh + mh*ql + ml*qh
      acc1 = __builtin_amdgcn_wmma_f32_16x16x32_bf16(false, A1h.v, false, Bh.v, (short)0, acc1, false, false);
      acc1 = __builtin_amdgcn_wmma_f32_16x16x32_bf16(false, A1h.v, false, Bl.v, (short)0, acc1, false, false);
      acc1 = __builtin_amdgcn_wmma_f32_16x16x32_bf16(false, A1l.v, false, Bh.v, (short)0, acc1, false, false);
      acc2 = __builtin_amdgcn_wmma_f32_16x16x32_bf16(false, A2h.v, false, Bh.v, (short)0, acc2, false, false);
      acc2 = __builtin_amdgcn_wmma_f32_16x16x32_bf16(false, A2h.v, false, Bl.v, (short)0, acc2, false, false);
      acc2 = __builtin_amdgcn_wmma_f32_16x16x32_bf16(false, A2l.v, false, Bh.v, (short)0, acc2, false, false);
    }

    // Rank by m2 - 2*dot (q2 per-query constant: order-invariant).
    const int r1 = c * 16 + hi8;
    #pragma unroll
    for (int r = 0; r < 8; ++r)
      topk_insert(dl9, il9, m2[r1 + r] - 2.0f * acc1[r], r1 + r);
    if (twoOk) {
      const int r2 = cB * 16 + hi8;
      #pragma unroll
      for (int r = 0; r < 8; ++r)
        topk_insert(dl9, il9, m2[r2 + r] - 2.0f * acc2[r], r2 + r);
    }
  }

  __syncthreads();
  merge_and_write(tid, qbase, seg, dl9, il9, md, mi, pd, pi);
}

// ---------------------------------------------------------------------------
// Kernel 1 (FALLBACK, small ws): on-the-fly bf16 split (round-1 kernel).
// ---------------------------------------------------------------------------
__global__ __launch_bounds__(128) void knn_fallback_kernel(
    const float* __restrict__ query, const float* __restrict__ memory,
    const float* __restrict__ m2, float* __restrict__ pd, int* __restrict__ pi) {
  __shared__ unsigned ldsqh[KSTEPS * 32 * 8];
  __shared__ unsigned ldsql[KSTEPS * 32 * 8];
  __shared__ float    md[128 * K_TOP];
  __shared__ int      mi[128 * K_TOP];

  const int tid   = threadIdx.x;
  const int lane  = tid & 31;
  const int wave  = tid >> 5;
  const int qbase = blockIdx.x * 16;
  const int seg   = blockIdx.y;

  stage_query_tile(query, qbase, tid, ldsqh, ldsql);
  __syncthreads();

  const int CPS = (NCHUNK + SEG - 1) / SEG;
  const int c0  = seg * CPS;
  const int c1  = (c0 + CPS < NCHUNK) ? (c0 + CPS) : NCHUNK;

  const float INF = __builtin_inff();
  float dl9[K_TOP]; int il9[K_TOP];
  #pragma unroll
  for (int j = 0; j < K_TOP; ++j) { dl9[j] = INF; il9[j] = -1; }

  const int arow_off = lane & 15;
  const int run0     = (lane >> 4) * 8;
  const int hi8      = (lane >> 4) << 3;

  for (int c = c0 + wave; c < c1; c += 4) {
    const int r0 = c * 16;
    const float* mrow_ptr = memory + (size_t)(r0 + arow_off) * D_DIM;
    v8f acc = {};
    #pragma unroll 2
    for (int ks = 0; ks < KSTEPS; ++ks) {
      const float* p0 = mrow_ptr + ks * 32 + run0;
      float4 a0 = *(const float4*)(p0 + 0);
      float4 a1 = *(const float4*)(p0 + 4);
      float4 b0 = *(const float4*)(p0 + 16);
      float4 b1 = *(const float4*)(p0 + 20);
      BF16x16 Ahf, Alf, Bh, Bl;
      split2(a0.x, a0.y, Ahf.u[0], Alf.u[0]);
      split2(a0.z, a0.w, Ahf.u[1], Alf.u[1]);
      split2(a1.x, a1.y, Ahf.u[2], Alf.u[2]);
      split2(a1.z, a1.w, Ahf.u[3], Alf.u[3]);
      split2(b0.x, b0.y, Ahf.u[4], Alf.u[4]);
      split2(b0.z, b0.w, Ahf.u[5], Alf.u[5]);
      split2(b1.x, b1.y, Ahf.u[6], Alf.u[6]);
      split2(b1.z, b1.w, Ahf.u[7], Alf.u[7]);
      const unsigned* qh = &ldsqh[(ks * 32 + lane) * 8];
      const unsigned* ql = &ldsql[(ks * 32 + lane) * 8];
      Bh.q[0] = *(const uint4*)(qh + 0);  Bh.q[1] = *(const uint4*)(qh + 4);
      Bl.q[0] = *(const uint4*)(ql + 0);  Bl.q[1] = *(const uint4*)(ql + 4);
      acc = __builtin_amdgcn_wmma_f32_16x16x32_bf16(false, Ahf.v, false, Bh.v, (short)0, acc, false, false);
      acc = __builtin_amdgcn_wmma_f32_16x16x32_bf16(false, Ahf.v, false, Bl.v, (short)0, acc, false, false);
      acc = __builtin_amdgcn_wmma_f32_16x16x32_bf16(false, Alf.v, false, Bh.v, (short)0, acc, false, false);
    }
    const int rb = r0 + hi8;
    #pragma unroll
    for (int r = 0; r < 8; ++r)
      topk_insert(dl9, il9, m2[rb + r] - 2.0f * acc[r], rb + r);
  }

  __syncthreads();
  merge_and_write(tid, qbase, seg, dl9, il9, md, mi, pd, pi);
}

// ---------------------------------------------------------------------------
// Kernel 2: merge SEG partial lists per query, add q2, sqrt, emit dist + idx.
// ---------------------------------------------------------------------------
__global__ __launch_bounds__(256) void finalize_kernel(
    const float* __restrict__ q2, const float* __restrict__ pd,
    const int* __restrict__ pi, float* __restrict__ outd, float* __restrict__ outi) {
  int q = blockIdx.x * 256 + threadIdx.x;
  if (q >= N_DIM) return;
  const float INF = __builtin_inff();
  float d[K_TOP]; int ix[K_TOP];
  #pragma unroll
  for (int j = 0; j < K_TOP; ++j) { d[j] = INF; ix[j] = -1; }
  for (int s = 0; s < SEG; ++s) {
    size_t o = ((size_t)q * SEG + s) * K_TOP;
    #pragma unroll
    for (int j = 0; j < K_TOP; ++j) topk_insert(d, ix, pd[o + j], pi[o + j]);
  }
  float qq = q2[q];
  #pragma unroll
  for (int j = 0; j < K_TOP; ++j) {
    float d2 = qq + d[j];
    outd[(size_t)q * K_TOP + j] = sqrtf(fmaxf(d2, 0.0f));
    outi[(size_t)q * K_TOP + j] = (float)ix[j];
  }
}

extern "C" void kernel_launch(void* const* d_in, const int* in_sizes, int n_in,
                              void* d_out, int out_size, void* d_ws, size_t ws_size,
                              hipStream_t stream) {
  const float* query  = (const float*)d_in[0];
  const float* memory = (const float*)d_in[1];
  // d_in[2] = k (==9), baked in at compile time.

  const size_t bytes_A  = NSLOTS * 8 * sizeof(unsigned);            // 76.8 MB each
  const size_t bytes_m2 = (size_t)M_DIM * 4;                        // 200 KB
  const size_t bytes_q2 = (size_t)N_DIM * 4;
  const size_t bytes_p  = (size_t)N_DIM * SEG * K_TOP * 4;          // 2.36 MB each
  const size_t need_fast = 2 * bytes_A + bytes_m2 + bytes_q2 + 2 * bytes_p;

  float* outd = (float*)d_out;
  float* outi = outd + (size_t)N_DIM * K_TOP;
  const int rows = M_DIM + N_DIM;

  if (ws_size >= need_fast) {
    char* p = (char*)d_ws;
    unsigned* Ah = (unsigned*)p;            p += bytes_A;
    unsigned* Al = (unsigned*)p;            p += bytes_A;
    float* m2 = (float*)p;                  p += bytes_m2;
    float* q2 = (float*)p;                  p += bytes_q2;
    float* pd = (float*)p;                  p += bytes_p;
    int*   pi = (int*)p;

    rownorm_kernel<<<(rows + 7) / 8, 256, 0, stream>>>(query, memory, m2, q2);
    operandize_kernel<<<(int)((NSLOTS + 255) / 256), 256, 0, stream>>>(memory, Ah, Al);
    dim3 g1(N_DIM / 16, SEG);
    knn_fast_kernel<<<g1, 128, 0, stream>>>(query, Ah, Al, m2, pd, pi);
    finalize_kernel<<<N_DIM / 256, 256, 0, stream>>>(q2, pd, pi, outd, outi);
  } else {
    float* ws = (float*)d_ws;
    float* m2 = ws;
    float* q2 = m2 + 50176;
    float* pd = q2 + N_DIM;
    int*   pi = (int*)(pd + (size_t)N_DIM * SEG * K_TOP);

    rownorm_kernel<<<(rows + 7) / 8, 256, 0, stream>>>(query, memory, m2, q2);
    dim3 g1(N_DIM / 16, SEG);
    knn_fallback_kernel<<<g1, 128, 0, stream>>>(query, memory, m2, pd, pi);
    finalize_kernel<<<N_DIM / 256, 256, 0, stream>>>(q2, pd, pi, outd, outi);
  }
}
